// CNN_56822417326399
// MI455X (gfx1250) — matrix operand order repaired and checked by reading, not compiled, not run on hardware
//
#include <hip/hip_runtime.h>
#include <hip/hip_bf16.h>

// ---------------------------------------------------------------------------
// Problem constants (from reference): B=128, L=512, D=300, V=50000, F=1000.
// Padded: K 300 -> 320 (10 WMMA K-steps of 32), F 1000 -> 1024 rows of W.
// ---------------------------------------------------------------------------
#define BB   128
#define LL   512
#define DD   300
#define KP   320        // padded K
#define VV   50000
#define FF   1000
#define FP   1024       // padded F (row count of staged W)
#define NFT  63         // ceil(1000/16) f-tiles

typedef __attribute__((ext_vector_type(16))) __bf16 v16bf;
typedef __attribute__((ext_vector_type(8)))  __bf16 v8bf;
typedef __attribute__((ext_vector_type(8)))  float  v8f;

// ---------------------------------------------------------------------------
// Prepass 1: emb f32 [V x 300] -> bf16 [V x 320], zero pad K=300..319.
// ---------------------------------------------------------------------------
__global__ void __launch_bounds__(256) cnn_cvt_emb(const float* __restrict__ emb,
                                                   __bf16* __restrict__ embb) {
    unsigned tid = blockIdx.x * 256u + threadIdx.x;
    if (tid >= (unsigned)(VV * KP)) return;
    unsigned r = tid / KP, k = tid - r * KP;
    float v = (k < DD) ? emb[(size_t)r * DD + k] : 0.0f;
    embb[tid] = (__bf16)v;
}

// ---------------------------------------------------------------------------
// Prepass 2: conv_w f32 [1000 x 300] -> bf16 [1024 x 320], zero padded.
// ---------------------------------------------------------------------------
__global__ void __launch_bounds__(256) cnn_cvt_w(const float* __restrict__ w,
                                                 __bf16* __restrict__ wtb) {
    unsigned tid = blockIdx.x * 256u + threadIdx.x;
    if (tid >= (unsigned)(FP * KP)) return;
    unsigned f = tid / KP, k = tid - f * KP;
    float v = (f < FF && k < DD) ? w[(size_t)f * DD + k] : 0.0f;
    wtb[tid] = (__bf16)v;
}

// ---------------------------------------------------------------------------
// Fused gathered-GEMM + bias + ReLU + max/argmax over t.
// Grid: (63 f-tiles, 128 batches), 256 threads = 8 wave32.
// Wave w handles t-tiles {w, w+8, w+16, w+24}; the WG covers all 512 t, so
// the max reduction completes inside the workgroup (no global atomics).
//
// All 10 K-step A-fragments are staged in registers before the WMMA chain so
// the compiler can issue one 20-load clause and wait *incrementally*
// (s_wait_loadcnt N) instead of draining LOADcnt before every WMMA.
//
// Keys: (f32 bits of relu(val)) << 32 | (511 - t)  -> max == argmax, first-
// occurrence tie-break (relu output >= 0 so the bit pattern is monotone).
// ---------------------------------------------------------------------------
__global__ void __launch_bounds__(256) cnn_gemm_max(
    const int*    __restrict__ inp,     // [B x L] token ids
    const __bf16* __restrict__ embb,    // [V x 320]
    const __bf16* __restrict__ wtb,     // [1024 x 320]
    const float*  __restrict__ conv_b,  // [1000]
    unsigned long long* __restrict__ keys) // [B x 1024]
{
    __shared__ unsigned long long red[8][16];

    const int ftile = blockIdx.x;           // 0..62
    const int b     = blockIdx.y;           // 0..127
    const int tid   = threadIdx.x;
    const int wave  = tid >> 5;
    const int lane  = tid & 31;
    const int g     = lane >> 4;            // lane half-group
    const int m     = lane & 15;            // A row (t) / B col (f) within tile
    const int f     = ftile * 16 + m;

    // Preload all 10 B fragments (reused by this wave across 4 t-tiles).
    // B fragment order used here (matches A): lane = column, 16 contiguous K starting at 16*g.
    v16bf bfrag[10];
    const __bf16* wrow = wtb + (size_t)f * KP + g * 16;
    #pragma unroll
    for (int ks = 0; ks < 10; ++ks)
        bfrag[ks] = *(const v16bf*)(wrow + ks * 32);

    const float bias = (f < FF) ? conv_b[f] : 0.0f;
    unsigned long long best = 0ull;

    int tok_next = inp[b * LL + wave * 16 + m];   // token for t-tile i=0

    #pragma unroll
    for (int i = 0; i < 4; ++i) {
        const int tbase = (wave + 8 * i) * 16;
        const int tok   = tok_next;
        // A fragment order used here (matches B): lane = row M; 16 contiguous K starting at 16*g.
        const __bf16* arow = embb + (size_t)tok * KP + g * 16;

        if (i < 3) {  // hoist next token id + prefetch its embedding row
            tok_next = inp[b * LL + (wave + 8 * (i + 1)) * 16 + m];
            __builtin_prefetch(embb + (size_t)tok_next * KP, 0, 3);
        }

        // Stage the whole 16x320 A strip for this t-tile (20 b128 loads).
        v8bf alo[10], ahi[10];
        #pragma unroll
        for (int ks = 0; ks < 10; ++ks) {
            alo[ks] = *(const v8bf*)(arow + ks * 32);
            ahi[ks] = *(const v8bf*)(arow + ks * 32 + 8);
        }

        v8f c = {};
        #pragma unroll
        for (int ks = 0; ks < 10; ++ks) {
            v16bf a = __builtin_shufflevector(alo[ks], ahi[ks],
                0, 1, 2, 3, 4, 5, 6, 7, 8, 9, 10, 11, 12, 13, 14, 15);
            c = __builtin_amdgcn_wmma_f32_16x16x32_bf16(
                    false, a, false, bfrag[ks], (short)0, c, false, false);
        }

        // C/D layout: VGPR r holds (M = r + 8g, N = m).
        #pragma unroll
        for (int r = 0; r < 8; ++r) {
            float v = fmaxf(c[r] + bias, 0.0f);
            int   t = tbase + r + 8 * g;
            unsigned long long key =
                ((unsigned long long)__float_as_uint(v) << 32) |
                (unsigned)(LL - 1 - t);
            best = key > best ? key : best;
        }
    }

    // Combine the two lane half-groups (same f, disjoint t halves).
    unsigned long long o = __shfl_xor(best, 16, 32);
    best = o > best ? o : best;
    if (lane < 16) red[wave][m] = best;
    __syncthreads();

    if (tid < 16) {
        unsigned long long fin = red[0][tid];
        #pragma unroll
        for (int w = 1; w < 8; ++w) {
            unsigned long long k2 = red[w][tid];
            fin = k2 > fin ? k2 : fin;
        }
        keys[(size_t)b * FP + ftile * 16 + tid] = fin;
    }
}

// ---------------------------------------------------------------------------
// Output init: token[b][l] = fc_b[1] - fc_b[0].
// ---------------------------------------------------------------------------
__global__ void __launch_bounds__(256) cnn_init_out(const float* __restrict__ fc_b,
                                                    float* __restrict__ out) {
    unsigned tid = blockIdx.x * 256u + threadIdx.x;
    if (tid < (unsigned)(BB * LL)) out[tid] = fc_b[1] - fc_b[0];
}

// ---------------------------------------------------------------------------
// Decode keys + scatter contributions: out[b][t*] += max * (fcw1 - fcw0).
// ---------------------------------------------------------------------------
__global__ void __launch_bounds__(256) cnn_scatter(
    const unsigned long long* __restrict__ keys,
    const float* __restrict__ fc_w,
    float* __restrict__ out)
{
    unsigned tid = blockIdx.x * 256u + threadIdx.x;   // 128 * 1024 threads
    unsigned b = tid >> 10, f = tid & 1023u;
    if (b >= BB || f >= FF) return;
    unsigned long long key = keys[tid];
    float    val = __uint_as_float((unsigned)(key >> 32));
    int      t   = (LL - 1) - (int)(unsigned)(key & 0xffffffffull);
    float contrib = val * (fc_w[FF + f] - fc_w[f]);
    atomicAdd(out + b * LL + t, contrib);
}

// ---------------------------------------------------------------------------
// Launcher. Inputs: 0=inp(int), 1=emb, 2=conv_w, 3=conv_b, 4=fc_w, 5=fc_b.
// Workspace: [0, 32MB) embb | [32MB, +640KB) wtb | then keys (1MB).
// ---------------------------------------------------------------------------
extern "C" void kernel_launch(void* const* d_in, const int* in_sizes, int n_in,
                              void* d_out, int out_size, void* d_ws, size_t ws_size,
                              hipStream_t stream) {
    const int*   inp    = (const int*)d_in[0];
    const float* emb    = (const float*)d_in[1];
    const float* conv_w = (const float*)d_in[2];
    const float* conv_b = (const float*)d_in[3];
    const float* fc_w   = (const float*)d_in[4];
    const float* fc_b   = (const float*)d_in[5];
    float*       out    = (float*)d_out;

    const size_t embb_bytes = (size_t)VV * KP * 2;        // 32,000,000
    const size_t wtb_bytes  = (size_t)FP * KP * 2;        //    655,360
    const size_t keys_bytes = (size_t)BB * FP * 8;        //  1,048,576
    if (ws_size < embb_bytes + wtb_bytes + keys_bytes) return;

    char* ws = (char*)d_ws;
    __bf16* embb = (__bf16*)ws;
    __bf16* wtb  = (__bf16*)(ws + embb_bytes);
    unsigned long long* keys = (unsigned long long*)(ws + embb_bytes + wtb_bytes);

    cnn_cvt_emb<<<(VV * KP + 255) / 256, 256, 0, stream>>>(emb, embb);
    cnn_cvt_w  <<<(FP * KP + 255) / 256, 256, 0, stream>>>(conv_w, wtb);

    dim3 grid(NFT, BB);
    cnn_gemm_max<<<grid, 256, 0, stream>>>(inp, embb, wtb, conv_b, keys);

    cnn_init_out<<<(BB * LL + 255) / 256, 256, 0, stream>>>(fc_b, out);
    cnn_scatter <<<(BB * FP + 255) / 256, 256, 0, stream>>>(keys, fc_w, out);
}
